// Senti_Attenuation_LSTM_89507118448944
// MI455X (gfx1250) — compile-verified, matrix operand order checked by reading, not run on hardware
//
#include <hip/hip_runtime.h>
#include <hip/hip_bf16.h>

typedef __attribute__((ext_vector_type(16))) __bf16 v16bf;
typedef __attribute__((ext_vector_type(8)))  float  v8f;
typedef __attribute__((ext_vector_type(4)))  int    v4i;

typedef __attribute__((address_space(1))) v4i as1_v4i;   // global v4i
typedef __attribute__((address_space(3))) v4i as3_v4i;   // LDS v4i

#if defined(__gfx1250__) && __has_builtin(__builtin_amdgcn_global_load_async_to_lds_b128)
#define USE_ASYNC_LDS 1
#else
#define USE_ASYNC_LDS 0
#endif

constexpr int B = 512, S = 256, D = 512, H = 512;
constexpr int K2 = 1024;          // [h_prev | x_t] concatenated reduction dim
constexpr int KC = 64;            // K chunk per staging round
constexpr int LDA = 72;           // LDS row stride in bf16 (144B: 16B-aligned, bank-skewed)

// ------------------------------------------------------------------
// Prep: transposed bf16 weight images (WMMA B-operand friendly).
// WcatT[n][k], n in [0,2048): gate g=n>>9, col c=n&511.
//   k <  512 -> V_g[k][c]   (recurrent part)
//   k >= 512 -> U_g[k-512][c] (input part)
// ------------------------------------------------------------------
__global__ void prep_wcat(const float* __restrict__ Vi, const float* __restrict__ Vf,
                          const float* __restrict__ Vc, const float* __restrict__ Vo,
                          const float* __restrict__ Ui, const float* __restrict__ Uf,
                          const float* __restrict__ Uc, const float* __restrict__ Uo,
                          __bf16* __restrict__ WcatT) {
  int idx = blockIdx.x * blockDim.x + threadIdx.x;    // = n*1024 + k
  int n = idx >> 10, k = idx & 1023;
  int g = n >> 9, c = n & 511;
  const float* V = (g == 0) ? Vi : (g == 1) ? Vf : (g == 2) ? Vc : Vo;
  const float* U = (g == 0) ? Ui : (g == 1) ? Uf : (g == 2) ? Uc : Uo;
  float v = (k < 512) ? V[k * H + c] : U[(k - 512) * H + c];
  WcatT[idx] = (__bf16)v;
}

__global__ void prep_we(const float* __restrict__ We, __bf16* __restrict__ WeT) {
  int idx = blockIdx.x * blockDim.x + threadIdx.x;    // = n*512 + k
  int n = idx >> 9, k = idx & 511;
  WeT[idx] = (__bf16)We[k * H + n];
}

__global__ void zero_state(__bf16* __restrict__ h0, __bf16* __restrict__ c0) {
  int idx = blockIdx.x * blockDim.x + threadIdx.x;
  h0[idx] = (__bf16)0.f;
  c0[idx] = (__bf16)0.f;
}

// ------------------------------------------------------------------
// Helpers
// ------------------------------------------------------------------
__device__ __forceinline__ void stage16B(const __bf16* gsrc, __bf16* ldst) {
#if USE_ASYNC_LDS
  __builtin_amdgcn_global_load_async_to_lds_b128((as1_v4i*)gsrc, (as3_v4i*)ldst, 0, 0);
#else
  *reinterpret_cast<uint4*>(ldst) = *reinterpret_cast<const uint4*>(gsrc);
#endif
}

__device__ __forceinline__ void wait_async() {
#if USE_ASYNC_LDS
#if __has_builtin(__builtin_amdgcn_s_wait_asynccnt)
  __builtin_amdgcn_s_wait_asynccnt(0);
#else
  asm volatile("s_wait_asynccnt 0x0" ::: "memory");
#endif
#endif
}

// WMMA 16-bit 16x32 fragment: lane row = lane&15, khalf = lane>>4;
// two contiguous 16B chunks at K = khalf*8 and 16+khalf*8.
__device__ __forceinline__ v16bf load_frag(const __bf16* rowp, int khalf) {
  union { v16bf v; uint4 q[2]; } u;
  u.q[0] = *reinterpret_cast<const uint4*>(rowp + (khalf << 3));
  u.q[1] = *reinterpret_cast<const uint4*>(rowp + 16 + (khalf << 3));
  return u.v;
}

__device__ __forceinline__ float fast_sigmoid(float x) {
  return __builtin_amdgcn_rcpf(1.0f + __builtin_amdgcn_exp2f(-1.44269504088896f * x));
}
__device__ __forceinline__ float fast_tanh(float x) {
  float e = __builtin_amdgcn_exp2f(2.88539008177793f * x);
  return (e - 1.0f) * __builtin_amdgcn_rcpf(e + 1.0f);
}

// ------------------------------------------------------------------
// One recurrence step.
// Block: 256 threads (8 waves). Tile: 64 batch-rows x 32 hidden-cols.
// Each wave owns one 16x16 C tile for 5 groups (i,f,c,o, c-tilde).
// Gates:   [h|x](64x1024) @ WcatT-cols (K=1024)
// c-tilde: c(64x512) @ WeT-cols (K=512), scaled by f_att in epilogue
// ------------------------------------------------------------------
__global__ __launch_bounds__(256)
void lstm_step(const float* __restrict__ x, const float* __restrict__ ts,
               const __bf16* __restrict__ WcatT, const __bf16* __restrict__ WeT,
               const float* __restrict__ bi, const float* __restrict__ bfg,
               const float* __restrict__ bc, const float* __restrict__ bo,
               const float* __restrict__ be,
               const __bf16* __restrict__ h_in, const __bf16* __restrict__ c_in,
               __bf16* __restrict__ h_out, __bf16* __restrict__ c_out,
               float* __restrict__ out, int t, int last) {
  __shared__ __bf16 Ah[64][LDA];   // [h|x] chunk, bf16
  __shared__ __bf16 Ae[64][LDA];   // c chunk, bf16
  __shared__ float  Fatt[64];      // per-row attenuation rate

  const int tid  = threadIdx.x;
  const int lane = tid & 31;
  const int w    = tid >> 5;        // wave 0..7
  const int mt   = w & 3;           // 16-row tile within 64 rows
  const int wn   = w >> 2;          // 16-col tile within 32 cols
  const int row0 = blockIdx.x * 64; // batch-row base
  const int col0 = blockIdx.y * 32; // hidden-col base

  // staging coordinates: each thread covers one row x 16 contiguous cols
  const int srow  = tid >> 2;          // 0..63
  const int seg16 = (tid & 3) << 4;    // 0,16,32,48

  const int arow  = mt * 16 + (lane & 15);
  const int khalf = lane >> 4;
  const int ncol  = col0 + wn * 16 + (lane & 15);   // global hidden col of this lane

  if (tid < 64) Fatt[tid] = ts[(size_t)(row0 + tid) * S + t];

  v8f acc_i = {}, acc_f = {}, acc_c = {}, acc_o = {}, acc_e = {};

  for (int kb = 0; kb < K2; kb += KC) {
    __syncthreads();
    // ---- stage A chunk into LDS ----
    if (kb < 512) {
      // recurrent half: raw bf16 copies (async global->LDS)
      const __bf16* hsrc = h_in + (size_t)(row0 + srow) * H + kb + seg16;
      stage16B(hsrc,     &Ah[srow][seg16]);
      stage16B(hsrc + 8, &Ah[srow][seg16 + 8]);
      const __bf16* csrc = c_in + (size_t)(row0 + srow) * H + kb + seg16;
      stage16B(csrc,     &Ae[srow][seg16]);
      stage16B(csrc + 8, &Ae[srow][seg16 + 8]);
      wait_async();
    } else {
      // input half: fp32 x -> bf16 conversion through VGPRs
      const float* src = x + ((size_t)(row0 + srow) * S + t) * D + (kb - 512) + seg16;
#pragma unroll
      for (int j = 0; j < 16; j += 4) {
        float4 f4 = *reinterpret_cast<const float4*>(src + j);
        __bf16* dst = &Ah[srow][seg16 + j];
        dst[0] = (__bf16)f4.x; dst[1] = (__bf16)f4.y;
        dst[2] = (__bf16)f4.z; dst[3] = (__bf16)f4.w;
      }
    }
    __syncthreads();

    // ---- WMMA: 2 sub-chunks of K=32, 4 gate tiles + c-tilde tile ----
#pragma unroll
    for (int half = 0; half < 2; ++half) {
      const int k0 = half << 5;
      v16bf a = load_frag(&Ah[arow][k0], khalf);
      {
        v16bf bfr = load_frag(WcatT + (size_t)(0 * 512 + ncol) * K2 + kb + k0, khalf);
        acc_i = __builtin_amdgcn_wmma_f32_16x16x32_bf16(false, a, false, bfr,
                                                        (short)0, acc_i, false, false);
      }
      {
        v16bf bfr = load_frag(WcatT + (size_t)(1 * 512 + ncol) * K2 + kb + k0, khalf);
        acc_f = __builtin_amdgcn_wmma_f32_16x16x32_bf16(false, a, false, bfr,
                                                        (short)0, acc_f, false, false);
      }
      {
        v16bf bfr = load_frag(WcatT + (size_t)(2 * 512 + ncol) * K2 + kb + k0, khalf);
        acc_c = __builtin_amdgcn_wmma_f32_16x16x32_bf16(false, a, false, bfr,
                                                        (short)0, acc_c, false, false);
      }
      {
        v16bf bfr = load_frag(WcatT + (size_t)(3 * 512 + ncol) * K2 + kb + k0, khalf);
        acc_o = __builtin_amdgcn_wmma_f32_16x16x32_bf16(false, a, false, bfr,
                                                        (short)0, acc_o, false, false);
      }
      if (kb < 512) {
        v16bf ae  = load_frag(&Ae[arow][k0], khalf);
        v16bf bfr = load_frag(WeT + (size_t)ncol * 512 + kb + k0, khalf);
        acc_e = __builtin_amdgcn_wmma_f32_16x16x32_bf16(false, ae, false, bfr,
                                                        (short)0, acc_e, false, false);
      }
    }
  }

  // ---- elementwise gate math + state update ----
  // C/D layout: lanes 0-15 N=lane M=r; lanes 16-31 N=lane-16 M=8+r.
  const float vbi = bi[ncol], vbf = bfg[ncol], vbc = bc[ncol];
  const float vbo = bo[ncol], vbe = be[ncol];
  const int rloc  = mt * 16 + ((lane >> 4) << 3);   // row within 64-row block
  const int rbase = row0 + rloc;
  const size_t OUT_HS = (size_t)B * S * H;
#pragma unroll
  for (int r = 0; r < 8; ++r) {
    float fa = Fatt[rloc + r];
    float it = fast_sigmoid(acc_i[r] + vbi);
    float ft = fast_sigmoid(acc_f[r] + vbf);
    float gt = fast_tanh(acc_c[r] + vbc);
    float ot = fast_sigmoid(acc_o[r] + vbo);
    float ct = fast_tanh(fa * acc_e[r] + vbe);   // f_att scaling moved out of GEMM
    float cn = ft * ct + it * gt;
    float hn = ot * fast_tanh(cn);
    int row = rbase + r;
    h_out[(size_t)row * H + ncol] = (__bf16)hn;
    c_out[(size_t)row * H + ncol] = (__bf16)cn;
    out[(size_t)row * S * H + (size_t)t * H + ncol] = hn;   // hidden_seq[b,t,:]
    if (last) {
      out[OUT_HS + (size_t)row * H + ncol] = hn;                 // h_t
      out[OUT_HS + (size_t)B * H + (size_t)row * H + ncol] = cn; // c_t
    }
  }
}

// ------------------------------------------------------------------
extern "C" void kernel_launch(void* const* d_in, const int* in_sizes, int n_in,
                              void* d_out, int out_size, void* d_ws, size_t ws_size,
                              hipStream_t stream) {
  (void)in_sizes; (void)n_in; (void)out_size; (void)ws_size;
  const float* x  = (const float*)d_in[0];
  const float* ts = (const float*)d_in[1];
  const float* We = (const float*)d_in[2];
  const float* be = (const float*)d_in[3];
  const float* Ui = (const float*)d_in[4];
  const float* Vi = (const float*)d_in[5];
  const float* bi = (const float*)d_in[6];
  const float* Uf = (const float*)d_in[7];
  const float* Vf = (const float*)d_in[8];
  const float* bf = (const float*)d_in[9];
  const float* Uc = (const float*)d_in[10];
  const float* Vc = (const float*)d_in[11];
  const float* bc = (const float*)d_in[12];
  const float* Uo = (const float*)d_in[13];
  const float* Vo = (const float*)d_in[14];
  const float* bo = (const float*)d_in[15];
  float* out = (float*)d_out;

  // workspace layout
  char* w = (char*)d_ws;
  __bf16* WcatT = (__bf16*)w;                                   // 2048*1024 bf16 = 4 MiB
  __bf16* WeT   = (__bf16*)(w + (size_t)2048 * 1024 * 2);       // 512*512 bf16 = 512 KiB
  __bf16* hbuf  = (__bf16*)(w + (size_t)2048 * 1024 * 2 + (size_t)512 * 512 * 2);
  __bf16* cbuf  = hbuf + 2 * (size_t)B * H;                     // 2x ping-pong each

  prep_wcat<<<(2048 * 1024) / 256, 256, 0, stream>>>(Vi, Vf, Vc, Vo, Ui, Uf, Uc, Uo, WcatT);
  prep_we<<<(512 * 512) / 256, 256, 0, stream>>>(We, WeT);
  zero_state<<<(B * H) / 256, 256, 0, stream>>>(hbuf, cbuf);    // zero buffer 0 (t=0 inputs)

  for (int t = 0; t < S; ++t) {
    const __bf16* hin  = hbuf + (size_t)(t & 1) * B * H;
    __bf16*       hout = hbuf + (size_t)((t + 1) & 1) * B * H;
    const __bf16* cin  = cbuf + (size_t)(t & 1) * B * H;
    __bf16*       cout = cbuf + (size_t)((t + 1) & 1) * B * H;
    lstm_step<<<dim3(8, 16), 256, 0, stream>>>(x, ts, WcatT, WeT,
                                               bi, bf, bc, bo, be,
                                               hin, cin, hout, cout,
                                               out, t, (t == S - 1) ? 1 : 0);
  }
}
